// WindowAttention_50792283242505
// MI455X (gfx1250) — matrix operand order repaired and checked by reading, not compiled, not run on hardware
//
#include <hip/hip_runtime.h>
#include <hip/hip_bf16.h>

// Fused Swin window attention for gfx1250 (MI455X), wave32 + WMMA.
// One workgroup (256 threads = 8 waves) per window of N=64 tokens, C=256.
// All intermediates (QK, V, P, O) stay in LDS (160 KB) -> HBM traffic is
// just x/zero_map in + out, the roofline minimum (~1.6 GB total).
// A tiny pre-pass converts the (L2-resident) f32 weights to f16 once per
// launch; streaming tensors use non-temporal hints so L2 keeps the weights.

typedef _Float16 f16;
typedef __attribute__((ext_vector_type(16))) _Float16 v16h;
typedef __attribute__((ext_vector_type(8)))  float    v8f;
typedef __attribute__((ext_vector_type(4)))  float    v4f;

#define N_TOK     64
#define DIM       256
#define NUM_HEADS 8
#define HEAD_DIM  32

__device__ __forceinline__ v8f wmma_f32(v16h a, v16h b, v8f c) {
  // v_wmma_f32_16x16x32_f16: D = A(16x32 f16) x B(32x16 f16) + C(16x16 f32)
  return __builtin_amdgcn_wmma_f32_16x16x32_f16(false, a, false, b,
                                                (short)0, c, false, false);
}

// A-operand (16x32 f16, M x K) from LDS. Layout: lane L holds row M=L&15;
// lanes 0-15: K=0..15, lanes 16-31: K=16..31 (matches B loaders).
__device__ __forceinline__ v16h load_A_lds(const f16* p, int ld) {
  const int l  = threadIdx.x & 31;
  const f16* rp = p + (l & 15) * ld;
  const int kb = (l < 16) ? 0 : 16;
  v16h a;
#pragma unroll
  for (int t = 0; t < 16; ++t) {
    const int kk = kb + t;
    a[t] = rp[kk];
  }
  return a;
}

// B-operand (32x16 f16, K x N) from LDS with generic strides:
// element (kk, n) at p[n*strideN + kk*strideK].
// Layout: lane L holds column N=L&15; lanes 0-15: K=0..15, lanes 16-31: K=16..31.
__device__ __forceinline__ v16h load_B_lds(const f16* p, int strideK, int strideN) {
  const int l  = threadIdx.x & 31;
  const f16* np = p + (l & 15) * strideN + ((l < 16) ? 0 : 16) * strideK;
  v16h b;
#pragma unroll
  for (int t = 0; t < 16; ++t) b[t] = np[t * strideK];
  return b;
}

// B-operand from a pre-converted f16 weight matrix W16[outF][K] (y = x W^T):
// B[kk][n] = W16[fbase+n][kbase+kk] -> each lane reads 16 CONTIGUOUS halves
// (32 B -> two global_load_b128 from L2-resident weights).
__device__ __forceinline__ v16h load_B_wgt16(const f16* __restrict__ W, int ldW,
                                             int fbase, int kbase) {
  const int l = threadIdx.x & 31;
  const f16* rp = W + (size_t)(fbase + (l & 15)) * ldW + kbase + ((l < 16) ? 0 : 16);
  v16h b;
#pragma unroll
  for (int t = 0; t < 16; ++t) b[t] = rp[t];
  return b;
}

// ---- pre-pass: f32 -> f16 weight conversion (runs once per launch) ----
__global__ __launch_bounds__(256)
void cvt_weights(const float* __restrict__ src, f16* __restrict__ dst, int n4) {
  const int i = blockIdx.x * blockDim.x + threadIdx.x;
  if (i < n4) {
    const v4f v = ((const v4f*)src)[i];
    dst[i * 4 + 0] = (f16)v.x;
    dst[i * 4 + 1] = (f16)v.y;
    dst[i * 4 + 2] = (f16)v.z;
    dst[i * 4 + 3] = (f16)v.w;
  }
}

__global__ __launch_bounds__(256)
void win_attn_fused(const float* __restrict__ x, const float* __restrict__ zmap,
                    const float* __restrict__ mask, const f16* __restrict__ Wv16,
                    const float* __restrict__ bv, const f16* __restrict__ Wqk16,
                    const float* __restrict__ rpb, const float* __restrict__ temperature,
                    const f16* __restrict__ Wp16, const float* __restrict__ bp,
                    float* __restrict__ out, int nW) {
  // 160 KB LDS: [ sX 32K | sZ 32K | sQK 64K | sV 32K ]; P aliases sX+sZ.
  __shared__ f16 smem[81920];
  f16* sX  = smem;           // x tile   [64 x 256]
  f16* sZ  = smem + 16384;   // zmap tile[64 x 256]
  f16* sP  = smem;           // softmax P, 8 heads x [64 x 64] (aliases sX/sZ)
  f16* sQK = smem + 32768;   // [64 x 512]: cols 0..255 = Q, 256..511 = K then O
  f16* sV  = smem + 65536;   // [64 x 256]

  const int b   = blockIdx.x;
  const int tid = threadIdx.x;
  const int w   = tid >> 5;          // wave id (0..7)
  const int l   = tid & 31;
  const int ln  = l & 15;            // N / column lane within C,D tiles
  const int hi8 = (l < 16) ? 0 : 8;  // C/D row offset for upper lane half

  // ---------- phase 0: stage x and zero_map into LDS as f16 ----------
  // Non-temporal: each window is read exactly once; keep L2 for weights.
  {
    const v4f* xg = (const v4f*)(x    + (size_t)b * (N_TOK * DIM));
    const v4f* zg = (const v4f*)(zmap + (size_t)b * (N_TOK * DIM));
#pragma unroll
    for (int it = 0; it < 16; ++it) {
      const int idx = tid + it * 256;  // 0..4095 float4s
      const v4f vx = __builtin_nontemporal_load(xg + idx);
      const v4f vz = __builtin_nontemporal_load(zg + idx);
      sX[idx * 4 + 0] = (f16)vx.x; sX[idx * 4 + 1] = (f16)vx.y;
      sX[idx * 4 + 2] = (f16)vx.z; sX[idx * 4 + 3] = (f16)vx.w;
      sZ[idx * 4 + 0] = (f16)vz.x; sZ[idx * 4 + 1] = (f16)vz.y;
      sZ[idx * 4 + 2] = (f16)vz.z; sZ[idx * 4 + 3] = (f16)vz.w;
    }
  }
  __syncthreads();

  // ---------- phase 1: QK = zmap @ Wqk^T -> sQK [64 x 512] ----------
  // 32 f-blocks of 16, 4 per wave; K=256 in 8 steps of 32.
  // Outer loop NOT unrolled: keeps one accumulator set live (<256 VGPRs).
#pragma unroll 1
  for (int fb = 0; fb < 4; ++fb) {
    const int fbase = (w * 4 + fb) * 16;
    if (l < 16) __builtin_prefetch(Wqk16 + (size_t)(fbase + l) * DIM, 0, 3);
    v8f acc[4] = {};
#pragma unroll
    for (int k = 0; k < 8; ++k) {
      const v16h B = load_B_wgt16(Wqk16, DIM, fbase, k * 32);
#pragma unroll
      for (int i = 0; i < 4; ++i) {
        const v16h A = load_A_lds(sZ + (16 * i) * DIM + k * 32, DIM);
        acc[i] = wmma_f32(A, B, acc[i]);
      }
    }
#pragma unroll
    for (int i = 0; i < 4; ++i)
#pragma unroll
      for (int r = 0; r < 8; ++r)
        sQK[(16 * i + r + hi8) * 512 + fbase + ln] = (f16)acc[i][r];
  }

  // ---------- phase 2: V = x @ Wv^T + bv -> sV [64 x 256] ----------
#pragma unroll 1
  for (int fb = 0; fb < 2; ++fb) {
    const int fbase = (w * 2 + fb) * 16;
    v8f acc[4] = {};
#pragma unroll
    for (int k = 0; k < 8; ++k) {
      const v16h B = load_B_wgt16(Wv16, DIM, fbase, k * 32);
#pragma unroll
      for (int i = 0; i < 4; ++i) {
        const v16h A = load_A_lds(sX + (16 * i) * DIM + k * 32, DIM);
        acc[i] = wmma_f32(A, B, acc[i]);
      }
    }
    const float bvv = bv[fbase + ln];
#pragma unroll
    for (int i = 0; i < 4; ++i)
#pragma unroll
      for (int r = 0; r < 8; ++r)
        sV[(16 * i + r + hi8) * DIM + fbase + ln] = (f16)(acc[i][r] + bvv);
  }
  __syncthreads();  // QK/V complete; sX/sZ dead -> reusable as P

  // ---------- phase 3: attention, wave w == head h ----------
  {
    const int h = w;
    const float st = temperature[h] * 0.17677669529663687f;  // temp * d^-0.5
    const float* mk = mask + (size_t)(b % nW) * (N_TOK * N_TOK);

    v16h Aq[4];
#pragma unroll
    for (int i = 0; i < 4; ++i)
      Aq[i] = load_A_lds(sQK + (16 * i) * 512 + h * HEAD_DIM, 512);

    v8f S[4][4];
#pragma unroll
    for (int j = 0; j < 4; ++j) {
      // K^T tile: B[kk][n] = QK[16j+n][256 + h*32 + kk]
      const v16h Bk = load_B_lds(sQK + (16 * j) * 512 + 256 + h * HEAD_DIM, 1, 512);
      const v8f z = {};
#pragma unroll
      for (int i = 0; i < 4; ++i) S[i][j] = wmma_f32(Aq[i], Bk, z);
    }

    // logits = S*scale*temp + rpb[h] + mask[b%nW]
#pragma unroll
    for (int i = 0; i < 4; ++i)
#pragma unroll
      for (int j = 0; j < 4; ++j)
#pragma unroll
        for (int r = 0; r < 8; ++r) {
          const int m  = 16 * i + r + hi8;
          const int mp = 16 * j + ln;
          const int dh = (m >> 3) - (mp >> 3) + 7;
          const int dw = (m & 7) - (mp & 7) + 7;
          const float bias = rpb[(dh * 15 + dw) * NUM_HEADS + h] + mk[m * 64 + mp];
          S[i][j][r] = S[i][j][r] * st + bias;
        }

    // row softmax: reduce across 4 j-tiles and the 16-lane half
    // (rows M and M+8 live in lane halves; xor masks 8/4/2/1 stay in-half)
#pragma unroll
    for (int i = 0; i < 4; ++i)
#pragma unroll
      for (int r = 0; r < 8; ++r) {
        float mx = fmaxf(fmaxf(S[i][0][r], S[i][1][r]),
                         fmaxf(S[i][2][r], S[i][3][r]));
#pragma unroll
        for (int off = 8; off >= 1; off >>= 1)
          mx = fmaxf(mx, __shfl_xor(mx, off, 32));
        float sum = 0.f;
#pragma unroll
        for (int j = 0; j < 4; ++j) {
          const float e = __expf(S[i][j][r] - mx);
          S[i][j][r] = e;
          sum += e;
        }
#pragma unroll
        for (int off = 8; off >= 1; off >>= 1)
          sum += __shfl_xor(sum, off, 32);
        const float rs = 1.f / sum;
#pragma unroll
        for (int j = 0; j < 4; ++j) S[i][j][r] *= rs;
      }

    // spill P (C/D layout) to LDS so it can be reloaded in A layout
    f16* Ph = sP + h * 4096;  // [64 x 64]
#pragma unroll
    for (int i = 0; i < 4; ++i)
#pragma unroll
      for (int j = 0; j < 4; ++j)
#pragma unroll
        for (int r = 0; r < 8; ++r)
          Ph[(16 * i + r + hi8) * 64 + 16 * j + ln] = (f16)S[i][j][r];

    // O_h = P @ V_h  [64x64]@[64x32]; overwrite this head's K columns in sQK
#pragma unroll
    for (int i = 0; i < 4; ++i)
#pragma unroll
      for (int jb = 0; jb < 2; ++jb) {
        v8f acc = {};
#pragma unroll
        for (int kt = 0; kt < 2; ++kt) {
          const v16h A = load_A_lds(Ph + (16 * i) * 64 + 32 * kt, 64);
          // B[kk][n] = V[32*kt + kk][h*32 + 16*jb + n]
          const v16h B = load_B_lds(sV + (32 * kt) * DIM + h * HEAD_DIM + 16 * jb,
                                    DIM, 1);
          acc = wmma_f32(A, B, acc);
        }
#pragma unroll
        for (int r = 0; r < 8; ++r)
          sQK[(16 * i + r + hi8) * 512 + 256 + h * HEAD_DIM + 16 * jb + ln] =
              (f16)acc[r];
      }
  }
  __syncthreads();  // all heads' O in sQK cols 256..511

  // ---------- phase 4: out = O @ Wp^T + bp -> global f32 (non-temporal) ----
  float* og = out + (size_t)b * (N_TOK * DIM);
#pragma unroll 1
  for (int fb = 0; fb < 2; ++fb) {
    const int fbase = (w * 2 + fb) * 16;
    if (l < 16) __builtin_prefetch(Wp16 + (size_t)(fbase + l) * DIM, 0, 3);
    v8f acc[4] = {};
#pragma unroll
    for (int k = 0; k < 8; ++k) {
      const v16h B = load_B_wgt16(Wp16, DIM, fbase, k * 32);
#pragma unroll
      for (int i = 0; i < 4; ++i) {
        const v16h A = load_A_lds(sQK + (16 * i) * 512 + 256 + k * 32, 512);
        acc[i] = wmma_f32(A, B, acc[i]);
      }
    }
    const float bpv = bp[fbase + ln];
#pragma unroll
    for (int i = 0; i < 4; ++i)
#pragma unroll
      for (int r = 0; r < 8; ++r)
        __builtin_nontemporal_store(acc[i][r] + bpv,
                                    og + (16 * i + r + hi8) * DIM + fbase + ln);
  }
}

extern "C" void kernel_launch(void* const* d_in, const int* in_sizes, int n_in,
                              void* d_out, int out_size, void* d_ws, size_t ws_size,
                              hipStream_t stream) {
  const float* x    = (const float*)d_in[0];
  const float* zm   = (const float*)d_in[1];
  const float* mask = (const float*)d_in[2];
  const float* Wv   = (const float*)d_in[3];
  const float* bv   = (const float*)d_in[4];
  const float* Wqk  = (const float*)d_in[5];
  const float* rpb  = (const float*)d_in[6];
  const float* temp = (const float*)d_in[7];
  const float* Wp   = (const float*)d_in[8];
  const float* bp   = (const float*)d_in[9];
  float* out = (float*)d_out;

  const int B  = in_sizes[0] / (N_TOK * DIM);          // 8192 windows
  const int nW = in_sizes[2] / (N_TOK * N_TOK);        // 1024 mask windows

  // f16 weight workspace layout (halves): Wqk16 | Wv16 | Wp16
  f16* Wqk16 = (f16*)d_ws;                 // 2C*C  = 131072 halves
  f16* Wv16  = Wqk16 + 2 * DIM * DIM;      //  C*C  =  65536 halves
  f16* Wp16  = Wv16 + DIM * DIM;           //  C*C  =  65536 halves

  // one-shot weight conversion (L2-resident, ~1 MB read total)
  {
    const int n4qk = (2 * DIM * DIM) / 4, n4 = (DIM * DIM) / 4;
    cvt_weights<<<dim3((n4qk + 255) / 256), dim3(256), 0, stream>>>(Wqk, Wqk16, n4qk);
    cvt_weights<<<dim3((n4 + 255) / 256),   dim3(256), 0, stream>>>(Wv,  Wv16,  n4);
    cvt_weights<<<dim3((n4 + 255) / 256),   dim3(256), 0, stream>>>(Wp,  Wp16,  n4);
  }

  win_attn_fused<<<dim3(B), dim3(256), 0, stream>>>(
      x, zm, mask, Wv16, bv, Wqk16, rpb, temp, Wp16, bp, out, nW);
}
